// TwoDimensionalSSM_55886114456152
// MI455X (gfx1250) — matrix operand order repaired and checked by reading, not compiled, not run on hardware
//
#include <hip/hip_runtime.h>
#include <hip/hip_bf16.h>
#include <math.h>

// ---------------------------------------------------------------------------
// 2-D SSM layer reformulated as 32 batched GEMMs (one per SSM channel):
//   out[s, b, e] = silu( sum_{s'} M_{e%32}[s,s'] * x[s',b,e] + x[s,b,e]*omega[e] )
// M_c combines all 4 flip-direction causal convolutions of the SSM impulse
// response into one dense 1024x1024 operator. GEMM runs in bf16 WMMA
// (v_wmma_f32_16x16x32_bf16) with f32 accumulation, double-buffered LDS and
// GLOBAL_LOAD_ASYNC_TO_LDS_B128 staging tracked by ASYNCcnt.
//
// Workspace layout (d_ws):
//   [0,       512KB)  : K field, f32  [128 heads][32][32]
//   [1MB,     65MB)   : M operators, bf16 [32 c][1024 s][1024 s']
//   [65MB,    97MB)   : packed X, bf16 [32 c][512 col][1024 s']   (col = b*16 + e/32)
// ---------------------------------------------------------------------------

typedef __bf16 bf16_t;
typedef __attribute__((ext_vector_type(16))) bf16_t v16bf;
typedef __attribute__((ext_vector_type(8)))  bf16_t v8bf;
typedef __attribute__((ext_vector_type(8)))  float  v8f;
typedef __attribute__((ext_vector_type(4)))  int    int4v;

#define L_SIDE 32
#define SEQ 1024
#define BSZ 32
#define EMBED 512
#define NCH 32            // SSM channels (kernel tiles with period 32 over E)
#define NCOL 512          // BSZ * (EMBED/NCH)

#if __has_builtin(__builtin_amdgcn_global_load_async_to_lds_b128)
#define HAVE_ASYNC_LDS 1
#else
#define HAVE_ASYNC_LDS 0
#endif

// Builtin takes int4 pointers in explicit address spaces:
//   param0: int4 addrspace(1)*  (global source, printed as "int4 __device__*")
//   param1: int4 addrspace(3)*  (LDS destination)
typedef __attribute__((address_space(1))) int4v g_int4;
typedef __attribute__((address_space(3))) int4v l_int4;

__device__ __forceinline__ float sigmoid_fast(float v) {
  // v_rcp_f32 instead of IEEE division (SiLU does not need RN division).
  return __builtin_amdgcn_rcpf(1.0f + __expf(-v));
}

__device__ __forceinline__ void async_copy16(const void* gp, void* lp) {
#if HAVE_ASYNC_LDS
  __builtin_amdgcn_global_load_async_to_lds_b128(
      (g_int4*)(uintptr_t)gp,
      (l_int4*)(uint32_t)(uintptr_t)lp,  // flat-LDS low 32 bits == LDS offset
      0, 0);
#else
  *(v8bf*)lp = *(const v8bf*)gp;
#endif
}

__device__ __forceinline__ void wait_async_all() {
#if HAVE_ASYNC_LDS
#if __has_builtin(__builtin_amdgcn_s_wait_asynccnt)
  __builtin_amdgcn_s_wait_asynccnt(0);
#else
  asm volatile("s_wait_asynccnt 0x0" ::: "memory");
#endif
#endif
}

// ---------------------------------------------------------------------------
// 1) SSM impulse response on the 32x32 grid: one thread per head (128 heads).
//    K[h][i][j], with the reference's boundary corrections baked in.
// ---------------------------------------------------------------------------
__global__ __launch_bounds__(128) void build_k(
    const float* __restrict__ A1p, const float* __restrict__ A2p,
    const float* __restrict__ A3p, const float* __restrict__ A4p,
    const float* __restrict__ B1p, const float* __restrict__ B2p,
    const float* __restrict__ C1p, const float* __restrict__ C2p,
    float* __restrict__ K) {
  const int h = threadIdx.x;  // 0..127
  float a1[2], a2[2], a3[2], a4[2], b1[2], b2[2], c1[2], c2[2];
#pragma unroll
  for (int n = 0; n < 2; ++n) {
    a1[n] = sigmoid_fast(A1p[h * 2 + n]);
    a2[n] = sigmoid_fast(A2p[h * 2 + n]);
    a3[n] = sigmoid_fast(A3p[h * 2 + n]);
    a4[n] = sigmoid_fast(A4p[h * 2 + n]);
    b1[n] = sigmoid_fast(B1p[h * 2 + n]);
    b2[n] = sigmoid_fast(B2p[h * 2 + n]);
    c1[n] = C1p[h * 2 + n];
    c2[n] = C2p[h * 2 + n];
  }
  const float scale = 0.7071067811865476f;  // sqrt(1/N), N=2

  float xh_up[2][L_SIDE];
  float xv_up[2][L_SIDE];
#pragma unroll
  for (int n = 0; n < 2; ++n)
    for (int j = 0; j < L_SIDE; ++j) { xh_up[n][j] = 0.f; xv_up[n][j] = 0.f; }

  for (int i = 0; i < L_SIDE; ++i) {
    float xv_row[2][L_SIDE];
    float xh_row[2][L_SIDE];
#pragma unroll
    for (int n = 0; n < 2; ++n) {
      for (int j = 0; j < L_SIDE; ++j) {
        float u = (i == 0 && j == 0) ? 1.f : 0.f;
        xv_row[n][j] = a3[n] * xh_up[n][j] + a4[n] * xv_up[n][j] + b2[n] * u;
      }
      float prev = 0.f;
      for (int j = 0; j < L_SIDE; ++j) {
        float u   = (i == 0 && j == 0) ? 1.f : 0.f;
        float xvl = (j == 0) ? 0.f : xv_row[n][j - 1];
        prev = a1[n] * prev + a2[n] * xvl + b1[n] * u;
        xh_row[n][j] = prev;
      }
    }
    for (int j = 0; j < L_SIDE; ++j) {
      float kv = scale * (c1[0] * xh_row[0][j] + c1[1] * xh_row[1][j] +
                          c2[0] * xv_row[0][j] + c2[1] * xv_row[1][j]);
      float bf = 1.f;
      if (i == 0) bf *= 2.f;
      if (j == 0) bf *= 2.f;
      if (i == 0 && j == 0) bf *= 0.25f;
      K[(h * L_SIDE + i) * L_SIDE + j] = kv * bf;
    }
#pragma unroll
    for (int n = 0; n < 2; ++n)
      for (int j = 0; j < L_SIDE; ++j) {
        xh_up[n][j] = xh_row[n][j];
        xv_up[n][j] = xv_row[n][j];
      }
  }
}

// ---------------------------------------------------------------------------
// 2) Materialize M_c[s, s'] (bf16): sum of the 4 flip-direction causal kernels.
// ---------------------------------------------------------------------------
__global__ __launch_bounds__(256) void build_M(const float* __restrict__ K,
                                               bf16_t* __restrict__ M) {
  size_t gid = (size_t)blockIdx.x * 256 + threadIdx.x;  // 32*1024*1024 total
  int sp = (int)(gid & 1023);
  int s  = (int)((gid >> 10) & 1023);
  int c  = (int)(gid >> 20);
  int i = s >> 5, j = s & 31;
  int ip = sp >> 5, jp = sp & 31;
  int di = i - ip, dj = j - jp;
  float v = 0.f;
  if (di >= 0 && dj >= 0) v += K[(((0 * NCH + c) * L_SIDE + di) * L_SIDE) + dj];
  if (di <= 0 && dj >= 0) v += K[(((1 * NCH + c) * L_SIDE - di) * L_SIDE) + dj];
  if (di >= 0 && dj <= 0) v += K[(((2 * NCH + c) * L_SIDE + di) * L_SIDE) - dj];
  if (di <= 0 && dj <= 0) v += K[(((3 * NCH + c) * L_SIDE - di) * L_SIDE) - dj];
  M[gid] = (bf16_t)v;
}

// ---------------------------------------------------------------------------
// 3) Pack x (S,B,E f32) -> Xt[c][col][s'] bf16, col = b*16 + e/32.
//    K dimension (s') innermost for coalesced GEMM B-tile loads.
// ---------------------------------------------------------------------------
__global__ __launch_bounds__(256) void pack_x(const float* __restrict__ x,
                                              bf16_t* __restrict__ Xt) {
  size_t gid = (size_t)blockIdx.x * 256 + threadIdx.x;  // 1024*32*512 total
  int e = (int)(gid & 511);
  int b = (int)((gid >> 9) & 31);
  int s = (int)(gid >> 14);
  int c = e & 31;
  int r = e >> 5;
  int col = b * 16 + r;
  Xt[((size_t)c * NCOL + col) * SEQ + s] = (bf16_t)x[gid];
}

// ---------------------------------------------------------------------------
// 4) Batched GEMM + fused residual/SiLU epilogue.
//    Block: 256 threads = 8 waves, 128x128 output tile, BK=32.
//    Double-buffered LDS; async global->LDS DMA overlaps WMMA.
//    Wave grid 4x2; each wave: 2x4 tiles of v_wmma_f32_16x16x32_bf16.
// ---------------------------------------------------------------------------
#define BM 128
#define BN 128
#define BK 32
#define KTILES (SEQ / BK)

__global__ __launch_bounds__(256) void gemm_silu(
    const bf16_t* __restrict__ M,     // [32][1024][1024]
    const bf16_t* __restrict__ Xt,    // [32][512][1024]
    const float*  __restrict__ x,     // [1024][32][512]
    const float*  __restrict__ omega, // [512]
    float*        __restrict__ out) { // [1024][32][512]
  __shared__ bf16_t Atile[2][BM][BK];  // M rows,   K contiguous
  __shared__ bf16_t Btile[2][BN][BK];  // cols (N), K contiguous

  const int c  = blockIdx.z;
  const int s0 = blockIdx.y * BM;
  const int n0 = blockIdx.x * BN;
  const int tid  = threadIdx.x;
  const int lane = tid & 31;
  const int wave = tid >> 5;   // 0..7
  const int wrow = wave >> 1;  // 0..3 -> 32 rows each
  const int wcol = wave & 1;   // 0..1 -> 64 cols each

  const bf16_t* Mc = M  + (size_t)c * SEQ * SEQ;
  const bf16_t* Xc = Xt + (size_t)c * NCOL * SEQ;

  // This thread's two 16B staging chunks: chunk -> (row 0..127, quarter 0..3)
  const int ch0  = tid * 2;
  const int row0 = ch0 >> 2, q0 = ch0 & 3;
  const int row1 = (ch0 + 1) >> 2, q1 = (ch0 + 1) & 3;

  auto load_tile = [&](int buf, int kt) {
    async_copy16(Mc + (size_t)(s0 + row0) * SEQ + kt + q0 * 8,
                 &Atile[buf][row0][q0 * 8]);
    async_copy16(Mc + (size_t)(s0 + row1) * SEQ + kt + q1 * 8,
                 &Atile[buf][row1][q1 * 8]);
    async_copy16(Xc + (size_t)(n0 + row0) * SEQ + kt + q0 * 8,
                 &Btile[buf][row0][q0 * 8]);
    async_copy16(Xc + (size_t)(n0 + row1) * SEQ + kt + q1 * 8,
                 &Btile[buf][row1][q1 * 8]);
  };

  v8f acc[2][4] = {};

  const int arow = lane & 15;
  const int ak0  = (lane < 16) ? 0 : 8;   // A: lanes 0-15 K{0-7,16-23}, 16-31 K{8-15,24-31}
  const int bk0  = (lane < 16) ? 0 : 8;   // B: lanes 0-15 K{0-7,16-23}, 16-31 K{8-15,24-31}

  load_tile(0, 0);
  wait_async_all();
  __syncthreads();

  for (int t = 0; t < KTILES; ++t) {
    const int p = t & 1;
    if (t + 1 < KTILES) load_tile(p ^ 1, (t + 1) * BK);  // overlap with WMMA

    // Assemble fragments per the CDNA5 16x32 A / 32x16 B per-lane layouts.
    v16bf afrag[2];
#pragma unroll
    for (int mi = 0; mi < 2; ++mi) {
      int r = wrow * 32 + mi * 16 + arow;
      union { v16bf v; v8bf h[2]; } u;
      u.h[0] = *(const v8bf*)(&Atile[p][r][ak0]);
      u.h[1] = *(const v8bf*)(&Atile[p][r][ak0 + 16]);
      afrag[mi] = u.v;
    }
    v16bf bfrag[4];
#pragma unroll
    for (int ni = 0; ni < 4; ++ni) {
      int col = wcol * 64 + ni * 16 + arow;
      union { v16bf v; v8bf h[2]; } u;
      u.h[0] = *(const v8bf*)(&Btile[p][col][bk0]);
      u.h[1] = *(const v8bf*)(&Btile[p][col][bk0 + 16]);
      bfrag[ni] = u.v;
    }

#pragma unroll
    for (int mi = 0; mi < 2; ++mi)
#pragma unroll
      for (int ni = 0; ni < 4; ++ni)
        acc[mi][ni] = __builtin_amdgcn_wmma_f32_16x16x32_bf16(
            false, afrag[mi], false, bfrag[ni], (short)0, acc[mi][ni],
            false, false);

    wait_async_all();   // next-buffer DMA done before anyone reads it
    __syncthreads();
  }

  // Epilogue: C/D layout -> lanes 0-15 rows v, lanes 16-31 rows v+8.
  const int lhalf = (lane >> 4) & 1;
#pragma unroll
  for (int ni = 0; ni < 4; ++ni) {
    int col = n0 + wcol * 64 + ni * 16 + (lane & 15);
    int b = col >> 4;
    int r = col & 15;
    int e = r * 32 + c;
    float om = omega[e];
#pragma unroll
    for (int mi = 0; mi < 2; ++mi) {
      int mbase = s0 + wrow * 32 + mi * 16 + lhalf * 8;
#pragma unroll
      for (int v = 0; v < 8; ++v) {
        int s = mbase + v;
        size_t idx = (size_t)s * (BSZ * EMBED) + (size_t)b * EMBED + e;
        float val = acc[mi][ni][v] + x[idx] * om;
        out[idx] = val * sigmoid_fast(val);  // SiLU
      }
    }
  }
}

// ---------------------------------------------------------------------------
extern "C" void kernel_launch(void* const* d_in, const int* in_sizes, int n_in,
                              void* d_out, int out_size, void* d_ws,
                              size_t ws_size, hipStream_t stream) {
  const float* x  = (const float*)d_in[0];
  const float* A1 = (const float*)d_in[1];
  const float* A2 = (const float*)d_in[2];
  const float* A3 = (const float*)d_in[3];
  const float* A4 = (const float*)d_in[4];
  const float* B1 = (const float*)d_in[5];
  const float* B2 = (const float*)d_in[6];
  const float* C1 = (const float*)d_in[7];
  const float* C2 = (const float*)d_in[8];
  const float* omega = (const float*)d_in[9];
  float* out = (float*)d_out;

  char* ws = (char*)d_ws;
  float*  Kf = (float*)ws;                                   // 512 KB
  bf16_t* M  = (bf16_t*)(ws + (size_t)(1u << 20));           // 64 MB
  bf16_t* Xt = (bf16_t*)(ws + (size_t)(1u << 20) + (size_t)NCH * SEQ * SEQ * 2);

  build_k<<<1, 128, 0, stream>>>(A1, A2, A3, A4, B1, B2, C1, C2, Kf);

  // 32 * 1024 * 1024 elements / 256 threads
  build_M<<<131072, 256, 0, stream>>>(Kf, M);

  // 1024 * 32 * 512 elements / 256 threads
  pack_x<<<65536, 256, 0, stream>>>(x, Xt);

  dim3 grid(NCOL / BN, SEQ / BM, NCH);  // (4, 8, 32)
  gemm_silu<<<grid, 256, 0, stream>>>(M, Xt, x, omega, out);
}
